// TaskAlignedAssigner_58145267253840
// MI455X (gfx1250) — matrix operand order repaired and checked by reading, not compiled, not run on hardware
//
#include <hip/hip_runtime.h>
#include <hip/hip_bf16.h>
#include <stdint.h>

#define BS 32
#define NA 8400
#define NG 40
#define NC 80
#define TOPKK 13
#define EPSF 1e-9f

// output layout (floats), concatenated in reference return order
#define OUT_L 0L
#define OUT_B 268800L
#define OUT_S 1344000L
#define OUT_F 22848000L

typedef __attribute__((ext_vector_type(16))) _Float16 v16h;
typedef __attribute__((ext_vector_type(8)))  float    v8f;

__device__ __forceinline__ float iou_f(float gx1,float gy1,float gx2,float gy2,
                                       float px1,float py1,float px2,float py2){
  float lx = fmaxf(gx1,px1), ly = fmaxf(gy1,py1);
  float rx = fminf(gx2,px2), ry = fminf(gy2,py2);
  float w = fmaxf(rx-lx,0.f), h = fmaxf(ry-ly,0.f);
  float ov = w*h;
  float a1 = (gx2-gx1)*(gy2-gy1);
  float a2 = (px2-px1)*(py2-py1);
  return ov / (a1 + a2 - ov + EPSF);
}

// ------------------------------------------------------------------
// Kernel A: one block per (b,g). Compute masked align metric for all
// 8400 anchors into LDS, take top-13 (jax top_k tie semantics:
// descending value, ties -> lowest index), emit mask_pos row.
// Also zero the per-(b,g) max tables used by kernel B's atomics.
// ------------------------------------------------------------------
__global__ __launch_bounds__(256) void kA_topk(
    const float* __restrict__ pd_scores, const float* __restrict__ pd_bboxes,
    const float* __restrict__ anc, const int* __restrict__ gt_labels,
    const float* __restrict__ gt_bboxes, const float* __restrict__ mask_gt,
    uint8_t* __restrict__ mask_pos, unsigned* __restrict__ pos_align,
    unsigned* __restrict__ pos_ovl)
{
  __shared__ float vals[NA];
  __shared__ float srv[256];
  __shared__ int   sri[256];
  __shared__ int   spick[TOPKK];

  int bg = blockIdx.x;
  int b = bg / NG;
  int tid = threadIdx.x;
  if (tid == 0) { pos_align[bg] = 0u; pos_ovl[bg] = 0u; }

  uint8_t* mrow = mask_pos + (size_t)bg * NA;
  float mg = mask_gt[bg];
  const float4 gb = reinterpret_cast<const float4*>(gt_bboxes)[bg];
  int lab = gt_labels[bg]; if (lab < 0 || lab >= NC) lab = 0;

  if (mg == 0.f) {                       // mask_gt zero: whole row is zero
    for (int a = tid; a < NA; a += 256) mrow[a] = 0;
    return;
  }

  const float4* pb = reinterpret_cast<const float4*>(pd_bboxes) + (size_t)b * NA;
  const float2* ap = reinterpret_cast<const float2*>(anc);
  const float*  sc = pd_scores + (size_t)b * NA * NC + lab;

  for (int a = tid; a < NA; a += 256) {
    mrow[a] = 0;
    float4 p = pb[a];
    float2 c = ap[a];
    float d0 = c.x - gb.x, d1 = c.y - gb.y, d2 = gb.z - c.x, d3 = gb.w - c.y;
    float md = fminf(fminf(d0, d1), fminf(d2, d3));
    float in = (md > EPSF) ? 1.f : 0.f;
    float iou = iou_f(gb.x, gb.y, gb.z, gb.w, p.x, p.y, p.z, p.w);
    float s  = sc[(size_t)a * NC];
    float i2 = iou * iou;
    vals[a] = s * (i2 * i2 * i2) * in;   // score^1 * iou^6 * mask_in_gts
  }
  __syncthreads();

  for (int t = 0; t < TOPKK; t++) {
    float bv = -1.f; int bi = 0x7fffffff;
    for (int a = tid; a < NA; a += 256) {
      float v = vals[a];
      if (v > bv) { bv = v; bi = a; }    // increasing a -> ties keep lowest
    }
    srv[tid] = bv; sri[tid] = bi;
    __syncthreads();
    for (int st = 128; st > 0; st >>= 1) {
      if (tid < st) {
        float ov2 = srv[tid + st]; int oi = sri[tid + st];
        if (ov2 > srv[tid] || (ov2 == srv[tid] && oi < sri[tid])) {
          srv[tid] = ov2; sri[tid] = oi;
        }
      }
      __syncthreads();
    }
    if (tid == 0) { int w = sri[0]; spick[t] = w; vals[w] = -2.f; }
    __syncthreads();
  }

  if (tid < TOPKK) {                     // mask_topk * mask_in_gts (mask_gt==1 here)
    int w = spick[tid];
    float2 c = ap[w];
    float d0 = c.x - gb.x, d1 = c.y - gb.y, d2 = gb.z - c.x, d3 = gb.w - c.y;
    float md = fminf(fminf(d0, d1), fminf(d2, d3));
    mrow[w] = (md > EPSF) ? 1 : 0;
  }
}

// ------------------------------------------------------------------
// Kernel B: one thread per (b,a). Stage GT boxes/labels into LDS via
// gfx1250 async-LDS loads. Resolve multi-GT anchors (argmax IoU over
// ALL gts, ties->first, per reference), write target_gt_idx / fg, and
// accumulate pos_align / pos_overlap maxima via bitwise atomicMax
// (valid: values are non-negative floats).
// ------------------------------------------------------------------
__global__ __launch_bounds__(256) void kB_resolve(
    const float* __restrict__ pd_scores, const float* __restrict__ pd_bboxes,
    const int* __restrict__ gt_labels, const float* __restrict__ gt_bboxes,
    const uint8_t* __restrict__ mask_pos,
    unsigned* __restrict__ pos_align, unsigned* __restrict__ pos_ovl,
    int* __restrict__ tgt_idx, int* __restrict__ fgm)
{
  __shared__ float s_gt[NG * 4];
  __shared__ int   s_lab[NG];

  int tid = threadIdx.x;
  const int blocksPerB = (NA + 255) / 256;
  int b = blockIdx.x / blocksPerB;
  int a = (blockIdx.x % blocksPerB) * 256 + tid;

  if (tid < NG) {                         // async copy: global -> LDS (ASYNCcnt)
    const float* gsrc = gt_bboxes + ((size_t)b * NG + tid) * 4;
    unsigned ld0 = (unsigned)(size_t)&s_gt[tid * 4];
    asm volatile("global_load_async_to_lds_b128 %0, %1, off"
                 :: "v"(ld0), "v"(gsrc) : "memory");
    const int* lsrc = gt_labels + (size_t)b * NG + tid;
    unsigned ld1 = (unsigned)(size_t)&s_lab[tid];
    asm volatile("global_load_async_to_lds_b32 %0, %1, off"
                 :: "v"(ld1), "v"(lsrc) : "memory");
  }
  asm volatile("s_wait_asynccnt 0" ::: "memory");
  __syncthreads();

  if (a >= NA) return;

  const uint8_t* mp = mask_pos + (size_t)b * NG * NA + a;
  int cnt = 0, first = -1;
  for (int g = 0; g < NG; g++) {
    int m = mp[(size_t)g * NA];
    if (m) { if (first < 0) first = g; cnt++; }
  }

  float4 p = reinterpret_cast<const float4*>(pd_bboxes)[(size_t)b * NA + a];
  int gsel, fg;
  if (cnt > 1) {
    float best = -1.f; int bi = 0;
    for (int g = 0; g < NG; g++) {
      float iou = iou_f(s_gt[g*4], s_gt[g*4+1], s_gt[g*4+2], s_gt[g*4+3],
                        p.x, p.y, p.z, p.w);
      if (iou > best) { best = iou; bi = g; }   // ties -> first (jnp.argmax)
    }
    gsel = bi; fg = 1;
  } else if (cnt == 1) { gsel = first; fg = 1; }
  else { gsel = 0; fg = 0; }

  size_t A = (size_t)b * NA + a;
  tgt_idx[A] = gsel;
  fgm[A] = fg;

  if (fg) {
    int lab = s_lab[gsel]; if (lab < 0 || lab >= NC) lab = 0;
    float s   = pd_scores[A * NC + lab];
    float iou = iou_f(s_gt[gsel*4], s_gt[gsel*4+1], s_gt[gsel*4+2], s_gt[gsel*4+3],
                      p.x, p.y, p.z, p.w);
    float i2 = iou * iou;
    float al = s * (i2 * i2 * i2);
    int bg = b * NG + gsel;
    atomicMax(&pos_align[bg], __float_as_uint(al));
    atomicMax(&pos_ovl[bg],  __float_as_uint(iou));
  }
}

// ------------------------------------------------------------------
// Kernel C: one wave32 per 16 anchors. Lanes 0..15 own one anchor each:
// compute norm, emit labels/bboxes/fg. Then the whole wave builds each
// 16x16 tile of target_scores with v_wmma_f32_16x16x32_f16:
//   A[m][k] = onehot(label_m within class tile), B = identity(16) ->
//   D = onehot tile in f32 accumulators, scaled per-row by norm.
// 5 class tiles cover all 80 classes, so every output element is
// produced by the matrix pipe (zeros included -> no separate memset).
// ------------------------------------------------------------------
__global__ __launch_bounds__(256) void kC_out(
    const float* __restrict__ pd_scores, const float* __restrict__ pd_bboxes,
    const int* __restrict__ gt_labels, const float* __restrict__ gt_bboxes,
    const unsigned* __restrict__ pos_align, const unsigned* __restrict__ pos_ovl,
    const int* __restrict__ tgt_idx, const int* __restrict__ fgm,
    float* __restrict__ out)
{
  int gtid = blockIdx.x * blockDim.x + threadIdx.x;
  int tile = gtid >> 5;                  // one wave32 per 16-anchor tile
  int lane = threadIdx.x & 31;
  long Abase = (long)tile * 16;          // NA % 16 == 0 -> tile never crosses b
  int b = (int)(Abase / NA);

  float norm = 0.f, fgf = 0.f; int lab = 0;

  if (lane < 16) {
    long A = Abase + lane;
    int a = (int)(A - (long)b * NA);
    int fg   = fgm[A];
    int gsel = tgt_idx[A];
    int bg = b * NG + gsel;
    float4 gb = reinterpret_cast<const float4*>(gt_bboxes)[bg];
    lab = gt_labels[bg]; if (lab < 0 || lab >= NC) lab = 0;
    if (fg) {
      float pa = __uint_as_float(pos_align[bg]);
      if (pa > EPSF) {
        float po = __uint_as_float(pos_ovl[bg]);
        float4 p = reinterpret_cast<const float4*>(pd_bboxes)[(size_t)b * NA + a];
        float s  = pd_scores[((size_t)b * NA + a) * NC + lab];
        float iou = iou_f(gb.x, gb.y, gb.z, gb.w, p.x, p.y, p.z, p.w);
        float i2 = iou * iou;
        norm = s * (i2 * i2 * i2) * po / pa;
        if (!(norm == norm)) norm = 0.f;           // nan_to_num
        norm = fminf(fmaxf(norm, 0.f), 1.f);       // clip 0..1
      }
    }
    fgf = fg ? 1.f : 0.f;
    out[OUT_L + A] = (float)lab;
    reinterpret_cast<float4*>(out + OUT_B)[A] = gb;
    out[OUT_F + A] = fgf;
  }

  // B fragment: identity over K<16 (32x16 f16 B: lanes 0-15 hold K=0..15
  // for column N=lane; lanes 16-31 hold K=16..31 -> zero).
  v16h bf;
  #pragma unroll
  for (int i = 0; i < 16; i++)
    bf[i] = (lane < 16 && i == lane) ? (_Float16)1.0f : (_Float16)0.0f;

  int m  = lane & 15;
  int lm = __shfl(lab, m, 32);           // label of anchor row m
  float* osc = out + OUT_S;

  #pragma unroll
  for (int ct = 0; ct < 5; ct++) {       // 5 x 16 classes = 80
    int ll = lm - ct * 16;               // label position inside this tile
    // A fragment (16x32 f16): lanes 0-15 hold K = p;
    // lanes 16-31 hold K = p + 16.
    v16h af;
    #pragma unroll
    for (int p = 0; p < 16; p++) {
      int K = p + ((lane >= 16) ? 16 : 0);
      af[p] = (K == ll) ? (_Float16)1.0f : (_Float16)0.0f;
    }
    v8f c = {};
    c = __builtin_amdgcn_wmma_f32_16x16x32_f16(
            /*neg_a=*/false, af, /*neg_b=*/false, bf,
            /*c_mod=*/(short)0, c, /*reuse_a=*/false, /*reuse_b=*/false);
    // D layout: VGPR r -> M = r (lanes 0-15) or 8+r (lanes 16-31); N = lane&15
    #pragma unroll
    for (int r = 0; r < 8; r++) {
      int M = r + ((lane >= 16) ? 8 : 0);
      float nm = __shfl(norm, M, 32);
      osc[(Abase + M) * NC + ct * 16 + (lane & 15)] = c[r] * nm;
    }
  }
}

extern "C" void kernel_launch(void* const* d_in, const int* in_sizes, int n_in,
                              void* d_out, int out_size, void* d_ws, size_t ws_size,
                              hipStream_t stream) {
  const float* pd_scores = (const float*)d_in[0];
  const float* pd_bboxes = (const float*)d_in[1];
  const float* anc       = (const float*)d_in[2];
  const int*   gt_labels = (const int*)d_in[3];
  const float* gt_bboxes = (const float*)d_in[4];
  const float* mask_gt   = (const float*)d_in[5];
  float* out = (float*)d_out;

  uint8_t* ws = (uint8_t*)d_ws;
  // workspace layout (all offsets 16B aligned):
  uint8_t*  mask_pos  = ws;                                    // 10,752,000 B
  unsigned* pos_align = (unsigned*)(ws + 10752000);            //      5,120 B
  unsigned* pos_ovl   = (unsigned*)(ws + 10752000 + 5120);     //      5,120 B
  int*      tgt       = (int*)(ws + 10752000 + 10240);         //  1,075,200 B
  int*      fgm       = (int*)(ws + 10752000 + 10240 + 1075200); // 1,075,200 B

  hipLaunchKernelGGL(kA_topk, dim3(BS * NG), dim3(256), 0, stream,
                     pd_scores, pd_bboxes, anc, gt_labels, gt_bboxes, mask_gt,
                     mask_pos, pos_align, pos_ovl);

  const int blocksPerB = (NA + 255) / 256;
  hipLaunchKernelGGL(kB_resolve, dim3(BS * blocksPerB), dim3(256), 0, stream,
                     pd_scores, pd_bboxes, gt_labels, gt_bboxes,
                     mask_pos, pos_align, pos_ovl, tgt, fgm);

  const int nTiles = (BS * NA) / 16;          // 16800 waves
  hipLaunchKernelGGL(kC_out, dim3((nTiles * 32) / 256), dim3(256), 0, stream,
                     pd_scores, pd_bboxes, gt_labels, gt_bboxes,
                     pos_align, pos_ovl, tgt, fgm, out);
}
